// SingleBandLSTM_16045997818518
// MI455X (gfx1250) — matrix operand order repaired and checked by reading, not compiled, not run on hardware
//
#include <hip/hip_runtime.h>
#include <math.h>

// ---------------------------------------------------------------------------
// 2-layer LSTM (B=64, IN=16, H=512, T=1024) for gfx1250 / MI455X.
// Persistent-kernel design: 16 WGs x 256 threads, each WG owns 32 h-channels.
// Recurrent GEMMs use v_wmma_f32_16x16x32_f16 with pre-swizzled fp16 weights.
// ---------------------------------------------------------------------------

typedef __attribute__((ext_vector_type(16))) _Float16 v16h;
typedef __attribute__((ext_vector_type(8)))  _Float16 v8h;
typedef __attribute__((ext_vector_type(8)))  float    v8f;

static constexpr int Bn   = 64;     // batch
static constexpr int INn  = 16;     // input features
static constexpr int Hn   = 512;    // hidden
static constexpr int G4   = 2048;   // 4*H gate width
static constexpr int NWG  = 16;     // persistent workgroups
static constexpr int HS   = 32;     // h-channels per WG (512/16)
static constexpr int NTH  = 256;    // threads per WG (8 waves)
static constexpr int KT   = 16;     // K tiles per GEMM (512/32)

// ---------------------------------------------------------------------------
// Fragment loaders (CDNA5 WMMA 16x16x32 f16 layouts, CDNA5 ISA)
// ---------------------------------------------------------------------------

union V16U { v16h v; v8h h[2]; };

// A fragment (16x32 f16): lane = 16*half + m; lane holds
//   halves j0..7  = row m, cols kbase + (half?8:0) + 0..7
//   halves j8..15 = row m, cols kbase + 16 + (half?8:0) + 0..7
// -> two aligned 16-byte loads straight from row-major fp16 h.
__device__ __forceinline__ v16h load_a_frag(const _Float16* __restrict__ h,
                                            int mbase, int kbase, int lane) {
  const _Float16* p = h + (size_t)(mbase + (lane & 15)) * Hn
                        + kbase + ((lane >> 4) << 3);
  V16U u;
  u.h[0] = *(const v8h*)(p);
  u.h[1] = *(const v8h*)(p + 16);
  return u.v;
}

// B fragments are pre-swizzled: frag[((tile)*32 + lane)*16 + j], 32B/lane.
__device__ __forceinline__ v16h load_b_frag(const _Float16* __restrict__ frag,
                                            int tile, int lane) {
  return *(const v16h*)(frag + ((size_t)tile * 32 + (size_t)lane) * 16);
}

__device__ __forceinline__ float sigf(float x) {
  return 1.0f / (1.0f + __expf(-x));
}

// ---------------------------------------------------------------------------
// Setup kernels (one-time, scalar — negligible vs. the 1024-step recurrence)
// ---------------------------------------------------------------------------

// out[] = b_out ; zero h0/h1 state buffers (parity 0) ; reset sync counter
__global__ void k_init(float* __restrict__ out, const float* __restrict__ b_out,
                       _Float16* __restrict__ h0, _Float16* __restrict__ h1,
                       unsigned* __restrict__ cnt, int out_n) {
  const int idx = blockIdx.x * blockDim.x + threadIdx.x;
  if (idx < out_n) out[idx] = b_out[0];
  if (idx < Bn * Hn) { h0[idx] = (_Float16)0.f; h1[idx] = (_Float16)0.f; }
  if (idx == 0) *cnt = 0u;
}

// h_in = x @ w_in^T + b_in    [64 x 512]
__global__ void k_hin(const float* __restrict__ x, const float* __restrict__ w_in,
                      const float* __restrict__ b_in, float* __restrict__ hin) {
  const int idx = blockIdx.x * blockDim.x + threadIdx.x;
  if (idx >= Bn * Hn) return;
  const int b = idx / Hn, h = idx % Hn;
  float s = b_in[h];
  for (int i = 0; i < INn; ++i) s += x[b * INn + i] * w_in[h * INn + i];
  hin[idx] = s;
}

// xg0 = h_in @ w_ih0^T + b_ih0 + b_hh0   [64 x 2048]  (time-invariant!)
__global__ void k_xg0(const float* __restrict__ hin, const float* __restrict__ w_ih0,
                      const float* __restrict__ b_ih0, const float* __restrict__ b_hh0,
                      float* __restrict__ xg0) {
  const int idx = blockIdx.x * blockDim.x + threadIdx.x;
  if (idx >= Bn * G4) return;
  const int b = idx / G4, g = idx % G4;
  float s = b_ih0[g] + b_hh0[g];
  const float* hr = hin + (size_t)b * Hn;
  const float* wr = w_ih0 + (size_t)g * Hn;
  for (int k = 0; k < Hn; ++k) s += hr[k] * wr[k];
  xg0[idx] = s;
}

__global__ void k_bias1(const float* __restrict__ b_ih1, const float* __restrict__ b_hh1,
                        float* __restrict__ bias1) {
  const int g = blockIdx.x * blockDim.x + threadIdx.x;
  if (g < G4) bias1[g] = b_ih1[g] + b_hh1[g];
}

// Swizzle a [2048 x 512] f32 weight matrix into fp16 WMMA B fragments.
// B(k,n) = W[n][k]. B frag layout (32x16 f16): lane = 16*half + n;
// lane holds halves j0..7 = K = ktile*32 + half*8 + j, j8..15 = K = ktile*32 + 16 + half*8 + (j-8), col n.
__global__ void k_pack(const float* __restrict__ w, _Float16* __restrict__ frag) {
  const int idx = blockIdx.x * blockDim.x + threadIdx.x;   // (tile, lane)
  if (idx >= (G4 / 16) * KT * 32) return;
  const int lane  = idx & 31;
  const int tile  = idx >> 5;
  const int ntile = tile >> 4;      // 0..127  (gate-col tile)
  const int ktile = tile & 15;      // 0..15   (K tile)
  const int nrow  = ntile * 16 + (lane & 15);            // gate row of W
  const int kbase = ktile * 32 + ((lane >> 4) << 3);     // +8 for upper half-wave
  _Float16* dst = frag + (size_t)idx * 16;
  const float* src = w + (size_t)nrow * Hn + kbase;
#pragma unroll
  for (int j = 0; j < 16; ++j) dst[j] = (_Float16)src[(j & 7) + ((j >> 3) << 4)];
}

// ---------------------------------------------------------------------------
// Persistent recurrence kernel
// ---------------------------------------------------------------------------

__device__ __forceinline__ void grid_sync(unsigned* __restrict__ cnt, unsigned target) {
  __threadfence();
  __syncthreads();
  if (threadIdx.x == 0) {
    atomicAdd(cnt, 1u);
    while (__hip_atomic_load(cnt, __ATOMIC_RELAXED, __HIP_MEMORY_SCOPE_AGENT) < target)
      __builtin_amdgcn_s_sleep(1);
  }
  __syncthreads();
}

__global__ void __launch_bounds__(NTH, 1)
k_lstm(const _Float16* __restrict__ frHH0,   // layer0 w_hh fragments
       const _Float16* __restrict__ frIH1,   // layer1 w_ih fragments
       const _Float16* __restrict__ frHH1,   // layer1 w_hh fragments
       const float*    __restrict__ xg0,     // [64 x 2048] time-invariant layer0 input gates
       const float*    __restrict__ bias1,   // [2048]
       const float*    __restrict__ w_out,   // [512]
       _Float16*       __restrict__ h0buf,   // [2][64][512] double-buffered
       _Float16*       __restrict__ h1buf,   // [2][64][512]
       float*          __restrict__ out,     // [B][T] (pre-seeded with b_out)
       unsigned*       __restrict__ cnt,
       const int*      __restrict__ seq) {
  __shared__ float lds_g[Bn * 128];   // gate slice [64][i|f|g|o x 32]  (32 KB)
  __shared__ float lds_c0[Bn * HS];   // layer0 cell state f32          (8 KB)
  __shared__ float lds_c1[Bn * HS];   // layer1 cell state f32          (8 KB)
  __shared__ float lds_wout[HS];

  const int wg   = blockIdx.x;
  const int tid  = threadIdx.x;
  const int wave = tid >> 5;          // 0..7
  const int lane = tid & 31;
  const int Tv   = seq[0];

  for (int i = tid; i < Bn * HS; i += NTH) { lds_c0[i] = 0.f; lds_c1[i] = 0.f; }
  if (tid < HS) lds_wout[tid] = w_out[wg * HS + tid];
  __syncthreads();

  // wave -> gate N-tile: gate block q = wave/2, sub = wave%2
  // global gate col base = q*512 + wg*32 + sub*16  ->  ntile = q*32 + wg*2 + sub
  const int ntile   = (wave >> 1) * 32 + wg * 2 + (wave & 1);
  const int halfsel = lane >> 4;
  const int ncol    = lane & 15;

  unsigned sync_no = 0;

  for (int t = 0; t < Tv; ++t) {
    const int cur = t & 1;
    const _Float16* h0r = h0buf + (size_t)cur       * (Bn * Hn);
    _Float16*       h0w = h0buf + (size_t)(cur ^ 1) * (Bn * Hn);
    const _Float16* h1r = h1buf + (size_t)cur       * (Bn * Hn);
    _Float16*       h1w = h1buf + (size_t)(cur ^ 1) * (Bn * Hn);

    // ---------------- Layer 0: gates = xg0 + h0[t-1] @ w_hh0^T ----------------
    {
      v8f acc[4] = {};
      for (int kt = 0; kt < KT; ++kt) {
        const v16h bf = load_b_frag(frHH0, ntile * KT + kt, lane);
        __builtin_prefetch(frHH0 + ((size_t)(ntile * KT + kt + 1) * 32 + lane) * 16, 0, 0);
#pragma unroll
        for (int mt = 0; mt < 4; ++mt) {
          const v16h af = load_a_frag(h0r, mt * 16, kt * 32, lane);
          acc[mt] = __builtin_amdgcn_wmma_f32_16x16x32_f16(
              false, af, false, bf, (short)0, acc[mt], false, false);
        }
      }
#pragma unroll
      for (int mt = 0; mt < 4; ++mt)
#pragma unroll
        for (int r = 0; r < 8; ++r) {
          const int mm   = mt * 16 + halfsel * 8 + r;
          const int gcol = ntile * 16 + ncol;
          lds_g[mm * 128 + wave * 16 + ncol] = acc[mt][r] + xg0[(size_t)mm * G4 + gcol];
        }
    }
    __syncthreads();

    // elementwise LSTM cell, layer 0 -> h0[t] (fp16, published globally)
    for (int e = tid; e < Bn * HS; e += NTH) {
      const int b = e >> 5, hc = e & 31;
      const float gi = lds_g[b * 128 + hc];
      const float gf = lds_g[b * 128 + 32 + hc];
      const float gc = lds_g[b * 128 + 64 + hc];
      const float go = lds_g[b * 128 + 96 + hc];
      const float c  = sigf(gf) * lds_c0[e] + sigf(gi) * tanhf(gc);
      lds_c0[e] = c;
      const float h = sigf(go) * tanhf(c);
      h0w[(size_t)b * Hn + wg * HS + hc] = (_Float16)h;
    }
    grid_sync(cnt, (++sync_no) * NWG);   // h0[t] now visible to all WGs

    // -------- Layer 1: gates = bias1 + h0[t] @ w_ih1^T + h1[t-1] @ w_hh1^T ----
    {
      v8f acc[4] = {};
      for (int kt = 0; kt < KT; ++kt) {
        const v16h bf = load_b_frag(frIH1, ntile * KT + kt, lane);
#pragma unroll
        for (int mt = 0; mt < 4; ++mt) {
          const v16h af = load_a_frag(h0w, mt * 16, kt * 32, lane);
          acc[mt] = __builtin_amdgcn_wmma_f32_16x16x32_f16(
              false, af, false, bf, (short)0, acc[mt], false, false);
        }
      }
      for (int kt = 0; kt < KT; ++kt) {
        const v16h bf = load_b_frag(frHH1, ntile * KT + kt, lane);
#pragma unroll
        for (int mt = 0; mt < 4; ++mt) {
          const v16h af = load_a_frag(h1r, mt * 16, kt * 32, lane);
          acc[mt] = __builtin_amdgcn_wmma_f32_16x16x32_f16(
              false, af, false, bf, (short)0, acc[mt], false, false);
        }
      }
#pragma unroll
      for (int mt = 0; mt < 4; ++mt)
#pragma unroll
        for (int r = 0; r < 8; ++r) {
          const int mm   = mt * 16 + halfsel * 8 + r;
          const int gcol = ntile * 16 + ncol;
          lds_g[mm * 128 + wave * 16 + ncol] = acc[mt][r] + bias1[gcol];
        }
    }
    __syncthreads();

    // elementwise LSTM cell, layer 1 -> h1[t] + partial output projection
    for (int e = tid; e < Bn * HS; e += NTH) {
      const int b = e >> 5, hc = e & 31;
      const float gi = lds_g[b * 128 + hc];
      const float gf = lds_g[b * 128 + 32 + hc];
      const float gc = lds_g[b * 128 + 64 + hc];
      const float go = lds_g[b * 128 + 96 + hc];
      const float c  = sigf(gf) * lds_c1[e] + sigf(gi) * tanhf(gc);
      lds_c1[e] = c;
      const float h = sigf(go) * tanhf(c);
      h1w[(size_t)b * Hn + wg * HS + hc] = (_Float16)h;
      atomicAdd(&out[(size_t)b * Tv + t], h * lds_wout[hc]);
    }
    grid_sync(cnt, (++sync_no) * NWG);   // h1[t] visible; lds_g reusable
  }
}

// ---------------------------------------------------------------------------
// Host launcher
// ---------------------------------------------------------------------------

extern "C" void kernel_launch(void* const* d_in, const int* in_sizes, int n_in,
                              void* d_out, int out_size, void* d_ws, size_t ws_size,
                              hipStream_t stream) {
  const float* x     = (const float*)d_in[0];
  const float* w_in  = (const float*)d_in[1];
  const float* b_in  = (const float*)d_in[2];
  const float* w_ih0 = (const float*)d_in[3];
  const float* w_hh0 = (const float*)d_in[4];
  const float* b_ih0 = (const float*)d_in[5];
  const float* b_hh0 = (const float*)d_in[6];
  const float* w_ih1 = (const float*)d_in[7];
  const float* w_hh1 = (const float*)d_in[8];
  const float* b_ih1 = (const float*)d_in[9];
  const float* b_hh1 = (const float*)d_in[10];
  const float* w_out = (const float*)d_in[11];
  const float* b_out = (const float*)d_in[12];
  const int*   seq   = (const int*)d_in[13];
  float* out = (float*)d_out;

  // workspace carve-up (256B-aligned regions)
  char* ws = (char*)d_ws;
  size_t off = 0;
  auto take = [&](size_t bytes) -> char* {
    char* p = ws + off;
    off = (off + bytes + 255) & ~(size_t)255;
    return p;
  };
  const size_t fragBytes = (size_t)(G4 / 16) * KT * 32 * 16 * sizeof(_Float16); // 2 MB
  _Float16* frHH0 = (_Float16*)take(fragBytes);
  _Float16* frIH1 = (_Float16*)take(fragBytes);
  _Float16* frHH1 = (_Float16*)take(fragBytes);
  float*    xg0   = (float*)take((size_t)Bn * G4 * sizeof(float));
  float*    bias1 = (float*)take((size_t)G4 * sizeof(float));
  float*    hin   = (float*)take((size_t)Bn * Hn * sizeof(float));
  _Float16* h0buf = (_Float16*)take((size_t)2 * Bn * Hn * sizeof(_Float16));
  _Float16* h1buf = (_Float16*)take((size_t)2 * Bn * Hn * sizeof(_Float16));
  unsigned* cnt   = (unsigned*)take(256);
  (void)ws_size; (void)n_in; (void)in_sizes;

  // one-time setup
  {
    const int n = out_size > Bn * Hn ? out_size : Bn * Hn;
    k_init<<<(n + NTH - 1) / NTH, NTH, 0, stream>>>(out, b_out, h0buf, h1buf, cnt, out_size);
  }
  k_hin<<<(Bn * Hn + NTH - 1) / NTH, NTH, 0, stream>>>(x, w_in, b_in, hin);
  k_xg0<<<(Bn * G4 + NTH - 1) / NTH, NTH, 0, stream>>>(hin, w_ih0, b_ih0, b_hh0, xg0);
  k_bias1<<<(G4 + NTH - 1) / NTH, NTH, 0, stream>>>(b_ih1, b_hh1, bias1);
  {
    const int n = (G4 / 16) * KT * 32;
    k_pack<<<(n + NTH - 1) / NTH, NTH, 0, stream>>>(w_hh0, frHH0);
    k_pack<<<(n + NTH - 1) / NTH, NTH, 0, stream>>>(w_ih1, frIH1);
    k_pack<<<(n + NTH - 1) / NTH, NTH, 0, stream>>>(w_hh1, frHH1);
  }

  // persistent recurrence
  k_lstm<<<NWG, NTH, 0, stream>>>(frHH0, frIH1, frHH1, xg0, bias1, w_out,
                                  h0buf, h1buf, out, cnt, seq);
}
